// Decoder_52810917872216
// MI455X (gfx1250) — compile-verified, matrix operand order checked by reading, not run on hardware
//
#include <hip/hip_runtime.h>
#include <math.h>

typedef __attribute__((ext_vector_type(16))) _Float16 v16h;
typedef __attribute__((ext_vector_type(8)))  _Float16 v8h;
typedef __attribute__((ext_vector_type(8)))  float    v8f;
typedef __attribute__((ext_vector_type(4)))  float    v4f;

#define DEV __device__ __forceinline__

DEV float sigm(float x) { return 1.f / (1.f + __expf(-x)); }

// Full 32-K fragment from f32 memory. p = row + k0 + kb (16B aligned).
// ISA 16-bit operand layout: f[0..7] = k {0..7}, f[8..15] = k {16..23} (lane<16),
// lanes>=16 shifted by kb=8.
DEV v16h frag32_f32(const float* __restrict__ p) {
  const v4f* q = (const v4f*)p;
  v4f a0 = q[0], a1 = q[1];
  const v4f* r = (const v4f*)(p + 16);
  v4f b0 = r[0], b1 = r[1];
  v16h f;
#pragma unroll
  for (int i = 0; i < 4; ++i) {
    f[i]      = (_Float16)a0[i];
    f[4 + i]  = (_Float16)a1[i];
    f[8 + i]  = (_Float16)b0[i];
    f[12 + i] = (_Float16)b1[i];
  }
  return f;
}

// Tail fragment covering 16 valid K values (K%32==16 case): upper 8 halves zero.
DEV v16h frag16_f32(const float* __restrict__ p) {
  const v4f* q = (const v4f*)p;
  v4f a0 = q[0], a1 = q[1];
  v16h f;
#pragma unroll
  for (int i = 0; i < 4; ++i) {
    f[i]      = (_Float16)a0[i];
    f[4 + i]  = (_Float16)a1[i];
    f[8 + i]  = (_Float16)0.f;
    f[12 + i] = (_Float16)0.f;
  }
  return f;
}

// C[m,n] = act( sum_k A[m*lda+k] * W[n*ldw+k] + bias[n] )
// One wave per block; each wave owns NT adjacent 16x16 C tiles (A fragment reuse).
// Requires: M%16==0, N%(16*NT)==0, K%16==0.
template <int NT>
__global__ __launch_bounds__(32) void gemm_wmma(
    const float* __restrict__ A, const float* __restrict__ W,
    const float* __restrict__ bias, float* __restrict__ C,
    int K, int lda, int ldw, int ldc, int act) {
  int lane = threadIdx.x & 31;
  int hi = lane >> 4, l15 = lane & 15, kb = hi * 8;
  int n0 = blockIdx.x * (16 * NT);
  int m0 = blockIdx.y * 16;
  const float* arow = A + (size_t)(m0 + l15) * lda + kb;
  const float* wrow = W + (size_t)(n0 + l15) * ldw + kb;
  size_t wstep = (size_t)16 * ldw;  // delta between adjacent N tiles

  v8f acc[NT];
#pragma unroll
  for (int nt = 0; nt < NT; ++nt) acc[nt] = (v8f){};

  int K32 = K & ~31;
  for (int k0 = 0; k0 < K32; k0 += 32) {
    v16h a = frag32_f32(arow + k0);
    if (k0 + 64 < K32) __builtin_prefetch(arow + k0 + 64, 0, 1);
#pragma unroll
    for (int nt = 0; nt < NT; ++nt) {
      v16h b = frag32_f32(wrow + nt * wstep + k0);
      acc[nt] = __builtin_amdgcn_wmma_f32_16x16x32_f16(false, a, false, b, (short)0,
                                                       acc[nt], false, false);
    }
  }
  if (K & 16) {
    v16h a = frag16_f32(arow + K32);
#pragma unroll
    for (int nt = 0; nt < NT; ++nt) {
      v16h b = frag16_f32(wrow + nt * wstep + K32);
      acc[nt] = __builtin_amdgcn_wmma_f32_16x16x32_f16(false, a, false, b, (short)0,
                                                       acc[nt], false, false);
    }
  }

#pragma unroll
  for (int nt = 0; nt < NT; ++nt) {
    int col = n0 + nt * 16 + l15;
    float bv = bias ? bias[col] : 0.f;
#pragma unroll
    for (int r = 0; r < 8; ++r) {
      int m = m0 + r + hi * 8;  // C/D layout: lanes>=16 hold M+8
      float v = acc[nt][r] + bv;
      if (act == 1) v = fmaxf(v, 0.f);
      C[(size_t)m * ldc + col] = v;
    }
  }
}

// Teacher forcing shift: frames_in[t] = (t==0) ? 0 : gtf[t-1].  Rows m = t*16+b.
__global__ void shift_frames(const float* __restrict__ gtf, float* __restrict__ fi) {
  int i = blockIdx.x * blockDim.x + threadIdx.x;
  if (i < 2048 * 80) fi[i] = (i < 16 * 80) ? 0.f : gtf[i - 16 * 80];
}

// LSTM1 with fresh zero state each step: q = sig(go)*tanh(sig(gi)*tanh(gg))
__global__ void lstm1_act(const float* __restrict__ g, float* __restrict__ q) {
  int i = blockIdx.x * blockDim.x + threadIdx.x;
  if (i < 2048 * 1024) {
    int m = i >> 10, j = i & 1023;
    const float* row = g + (size_t)m * 4096;
    float gi = row[j], gg = row[2048 + j], go = row[3072 + j];
    float c = sigm(gi) * tanhf(gg);
    q[i] = sigm(go) * tanhf(c);
  }
}

__global__ void cvt_f16(const float* __restrict__ s, _Float16* __restrict__ d, int n) {
  int i = blockIdx.x * blockDim.x + threadIdx.x;
  if (i < n) d[i] = (_Float16)s[i];
}

// Location-sensitive attention scan, one block per batch element.
__global__ __launch_bounds__(256) void attention_kernel(
    const float* __restrict__ enc,    // [192,16,512]
    const float* __restrict__ keys,   // [(s*16+b), 128]
    const float* __restrict__ qp,     // [(t*16+b), 128]
    const float* __restrict__ convw,  // [32,31]
    const float* __restrict__ wloc,   // [128,32]
    const float* __restrict__ vvec,   // [128]
    float* __restrict__ x2,           // [2048,768] -> cols 256..767
    float* __restrict__ dec) {        // [2048,1536] -> cols 1024..1535
  __shared__ float s_align[192];
  __shared__ float s_loc[32 * 192];
  __shared__ float s_wloc[128 * 32];
  __shared__ float s_cw[32 * 31];
  __shared__ float s_v[128];
  __shared__ float s_q[128];
  __shared__ float s_e[192];
  __shared__ float s_inv[1];
  int b = blockIdx.x;
  int tid = threadIdx.x;
  for (int i = tid; i < 128 * 32; i += 256) s_wloc[i] = wloc[i];
  for (int i = tid; i < 32 * 31; i += 256) s_cw[i] = convw[i];
  if (tid < 128) s_v[tid] = vvec[tid];
  if (tid < 192) s_align[tid] = 0.f;
  __syncthreads();
  for (int t = 0; t < 128; ++t) {
    if (tid < 128) s_q[tid] = qp[(size_t)(t * 16 + b) * 128 + tid];
    // location conv of previous alignment (pad 15, kernel 31)
    for (int idx = tid; idx < 32 * 192; idx += 256) {
      int f = idx / 192, s = idx - f * 192;
      float a = 0.f;
#pragma unroll
      for (int j = 0; j < 31; ++j) {
        int sp = s + j - 15;
        if (sp >= 0 && sp < 192) a += s_align[sp] * s_cw[f * 31 + j];
      }
      s_loc[f * 192 + s] = a;
    }
    __syncthreads();
    if (tid < 192) {
      const float* krow = keys + (size_t)(tid * 16 + b) * 128;
      float acc = 0.f;
      for (int a = 0; a < 128; ++a) {
        float lf = 0.f;
#pragma unroll 8
        for (int f = 0; f < 32; ++f) lf += s_loc[f * 192 + tid] * s_wloc[a * 32 + f];
        acc += s_v[a] * tanhf(krow[a] + lf + s_q[a]);
      }
      s_e[tid] = acc;
    }
    __syncthreads();
    if (tid == 0) {
      float mx = -1e30f;
      for (int s = 0; s < 192; ++s) mx = fmaxf(mx, s_e[s]);
      float sm = 0.f;
      for (int s = 0; s < 192; ++s) { float ex = __expf(s_e[s] - mx); s_e[s] = ex; sm += ex; }
      s_inv[0] = 1.f / sm;
    }
    __syncthreads();
    if (tid < 192) s_align[tid] = s_e[tid] * s_inv[0];
    __syncthreads();
    // ctx[h] = sum_s align[s] * enc[s,b,h]; each thread owns h and h+256
#pragma unroll
    for (int hh = 0; hh < 2; ++hh) {
      int h = tid + hh * 256;
      float acc = 0.f;
      for (int s = 0; s < 192; ++s) acc += s_align[s] * enc[(size_t)(s * 16 + b) * 512 + h];
      x2[(size_t)(t * 16 + b) * 768 + 256 + h] = acc;
      dec[(size_t)(t * 16 + b) * 1536 + 1024 + h] = acc;
    }
    __syncthreads();
  }
}

// LSTM2 recurrence: single 32-wave block; gates = G[t] + h@Whh^T via WMMA.
// Each wave owns 2 hidden-unit tiles (j0 = wv*32 .. +31); the four gate tiles of a
// j-tile live at constant byte offsets from ONE base pointer (jt*32KB, gate*2MB),
// so every B-operand load is a base+imm global_load_b128 -> no address spills.
__global__ __launch_bounds__(1024, 1) void lstm2_kernel(
    const float* __restrict__ G,     // [2048,4096] = x2@Wih^T + b
    const _Float16* __restrict__ Wh, // [4096,1024] f16
    float* __restrict__ dec) {       // [2048,1536], writes cols 0..1023
  extern __shared__ char smem[];
  _Float16* hbuf = (_Float16*)smem;               // 16*1024 halves (32KB)
  float* cbuf = (float*)(smem + 16 * 1024 * 2);   // 16*1024 floats (64KB)
  int tid = threadIdx.x;
  int lane = tid & 31, wv = tid >> 5;
  int hi = lane >> 4, l15 = lane & 15, kb = hi * 8;
  for (int i = tid; i < 16 * 1024; i += 1024) { hbuf[i] = (_Float16)0.f; cbuf[i] = 0.f; }
  __syncthreads();

  const _Float16* hrow = hbuf + l15 * 1024 + kb;                       // LDS A row
  const _Float16* wbase = Wh + (size_t)(wv * 32 + l15) * 1024 + kb;    // B base

  for (int t = 0; t < 128; ++t) {
    // acc init from precomputed input gates: one base, all-immediate offsets
    const float* gbase = G + (size_t)(t * 16 + hi * 8) * 4096 + wv * 32 + l15;
    v8f acc[2][4];
#pragma unroll
    for (int jt = 0; jt < 2; ++jt)
#pragma unroll
      for (int gi = 0; gi < 4; ++gi) {
        v8f a;
#pragma unroll
        for (int r = 0; r < 8; ++r) a[r] = gbase[r * 4096 + jt * 16 + gi * 1024];
        acc[jt][gi] = a;
      }

    for (int k0 = 0; k0 < 1024; k0 += 32) {
      v16h af;
      {
        v8h x = *(const v8h*)(hrow + k0);
        v8h y = *(const v8h*)(hrow + k0 + 16);
#pragma unroll
        for (int i = 0; i < 8; ++i) { af[i] = x[i]; af[8 + i] = y[i]; }
      }
      const _Float16* wk0 = wbase + k0;
#pragma unroll
      for (int jt = 0; jt < 2; ++jt)
#pragma unroll
        for (int gi = 0; gi < 4; ++gi) {
          const _Float16* wp = wk0 + jt * (16 * 1024) + gi * (1024 * 1024);
          v8h x = *(const v8h*)(wp);
          v8h y = *(const v8h*)(wp + 16);
          v16h bf;
#pragma unroll
          for (int i = 0; i < 8; ++i) { bf[i] = x[i]; bf[8 + i] = y[i]; }
          acc[jt][gi] = __builtin_amdgcn_wmma_f32_16x16x32_f16(
              false, af, false, bf, (short)0, acc[jt][gi], false, false);
        }
    }
    __syncthreads();  // all hbuf reads done before updates
#pragma unroll
    for (int jt = 0; jt < 2; ++jt) {
      int j = wv * 32 + jt * 16 + l15;
#pragma unroll
      for (int r = 0; r < 8; ++r) {
        int m = r + hi * 8;
        float iv = acc[jt][0][r], fv = acc[jt][1][r];
        float gv = acc[jt][2][r], ov = acc[jt][3][r];
        float co = cbuf[m * 1024 + j];
        float cn = sigm(fv) * co + sigm(iv) * tanhf(gv);
        float hn = sigm(ov) * tanhf(cn);
        cbuf[m * 1024 + j] = cn;
        hbuf[m * 1024 + j] = (_Float16)hn;
        dec[(size_t)(t * 16 + m) * 1536 + j] = hn;
      }
    }
    __syncthreads();
  }
}

// im2col for conv1d k=5 pad=2 over rows m = t*16+b:  dst[m, ci*5+dk] = src[(t+dk-2)*16+b, ci]
__global__ void im2col(const float* __restrict__ src, float* __restrict__ dst, int C) {
  int i = blockIdx.x * blockDim.x + threadIdx.x;
  int K2 = C * 5, total = 2048 * K2;
  if (i < total) {
    int m = i / K2, r = i - m * K2;
    int ci = r / 5, dk = r - ci * 5;
    int t = m >> 4, b = m & 15, ts = t + dk - 2;
    dst[i] = (ts >= 0 && ts < 128) ? src[(size_t)(ts * 16 + b) * C + ci] : 0.f;
  }
}

// Training-mode BN stats over all 2048 (b,t) samples; fold gamma/beta into scale/shift.
__global__ __launch_bounds__(256) void bn_stats(const float* __restrict__ Y, int N,
                                                const float* __restrict__ gamma,
                                                const float* __restrict__ beta,
                                                float* __restrict__ scale,
                                                float* __restrict__ shift) {
  int co = blockIdx.x;
  __shared__ float s1[256], s2[256];
  float a = 0.f, b2 = 0.f;
  for (int m = threadIdx.x; m < 2048; m += 256) {
    float v = Y[(size_t)m * N + co];
    a += v; b2 += v * v;
  }
  s1[threadIdx.x] = a; s2[threadIdx.x] = b2;
  __syncthreads();
  for (int o = 128; o > 0; o >>= 1) {
    if (threadIdx.x < o) { s1[threadIdx.x] += s1[threadIdx.x + o]; s2[threadIdx.x] += s2[threadIdx.x + o]; }
    __syncthreads();
  }
  if (threadIdx.x == 0) {
    float mean = s1[0] * (1.f / 2048.f);
    float var = s2[0] * (1.f / 2048.f) - mean * mean;
    float sc = gamma[co] * rsqrtf(var + 1e-5f);
    scale[co] = sc;
    shift[co] = beta[co] - mean * sc;
  }
}

__global__ void bn_apply(float* __restrict__ Y, int N, const float* __restrict__ scale,
                         const float* __restrict__ shift, int do_tanh) {
  int i = blockIdx.x * blockDim.x + threadIdx.x;
  if (i < 2048 * N) {
    int co = i % N;
    float v = scale[co] * Y[i] + shift[co];
    Y[i] = do_tanh ? tanhf(v) : v;
  }
}

__global__ void residual_out(const float* __restrict__ frames, const float* __restrict__ R,
                             const float* __restrict__ scale, const float* __restrict__ shift,
                             float* __restrict__ out2) {
  int i = blockIdx.x * blockDim.x + threadIdx.x;
  if (i < 2048 * 80) {
    int co = i % 80;
    out2[i] = frames[i] + scale[co] * R[i] + shift[co];
  }
}

extern "C" void kernel_launch(void* const* d_in, const int* in_sizes, int n_in,
                              void* d_out, int out_size, void* d_ws, size_t ws_size,
                              hipStream_t stream) {
  (void)in_sizes; (void)n_in; (void)out_size; (void)ws_size;
  const float* enc    = (const float*)d_in[0];   // [192,16,512]
  const float* gtf    = (const float*)d_in[1];   // [128,16,80]
  const float* pre_w1 = (const float*)d_in[2];
  const float* pre_b1 = (const float*)d_in[3];
  const float* pre_w2 = (const float*)d_in[4];
  const float* pre_b2 = (const float*)d_in[5];
  const float* l1_wih = (const float*)d_in[6];   // [4096,768] (only cols <256 matter)
  const float* l1_b   = (const float*)d_in[8];
  const float* l2_wih = (const float*)d_in[9];
  const float* l2_whh = (const float*)d_in[10];
  const float* l2_b   = (const float*)d_in[11];
  const float* wq     = (const float*)d_in[12];
  const float* wk     = (const float*)d_in[13];
  const float* locw   = (const float*)d_in[14];
  const float* wloc   = (const float*)d_in[15];
  const float* vvec   = (const float*)d_in[16];
  const float* out_w  = (const float*)d_in[17];
  const float* out_b  = (const float*)d_in[18];
  const float* pnw[5] = {(const float*)d_in[19], (const float*)d_in[22],
                         (const float*)d_in[25], (const float*)d_in[28],
                         (const float*)d_in[31]};
  const float* png[5] = {(const float*)d_in[20], (const float*)d_in[23],
                         (const float*)d_in[26], (const float*)d_in[29],
                         (const float*)d_in[32]};
  const float* pnb[5] = {(const float*)d_in[21], (const float*)d_in[24],
                         (const float*)d_in[27], (const float*)d_in[30],
                         (const float*)d_in[33]};

  float* W = (float*)d_ws;
  size_t o = 0;
  float* BIG  = W + o; o += 2048ull * 4096;  // gates1 -> G2 -> im2col
  float* X2   = W + o; o += 2048ull * 768;   // [pre | ctx]
  float* H1   = W + o; o += 2048ull * 256;
  float* Q    = W + o; o += 2048ull * 1024;
  float* KEYS = W + o; o += 3072ull * 128;
  float* QP   = W + o; o += 2048ull * 128;
  float* DEC  = W + o; o += 2048ull * 1536;  // [h2 | ctx]
  float* FI   = W + o; o += 2048ull * 80;
  float* P    = W + o; o += 2048ull * 512;
  float* R    = W + o; o += 2048ull * 80;
  float* SC   = W + o; o += 512;
  float* SH   = W + o; o += 512;
  _Float16* WHH = (_Float16*)(W + o);        // [4096,1024] f16

  float* frames = (float*)d_out;
  float* out2 = frames + 2048 * 80;
  dim3 b32(32);

  shift_frames<<<640, 256, 0, stream>>>(gtf, FI);
  // PreNet
  gemm_wmma<4><<<dim3(4, 128), b32, 0, stream>>>(FI, pre_w1, pre_b1, H1, 80, 80, 80, 256, 1);
  gemm_wmma<4><<<dim3(4, 128), b32, 0, stream>>>(H1, pre_w2, pre_b2, X2, 256, 256, 256, 768, 1);
  // LSTM1 gates (zero context + zero state => batched GEMM, K=256 of wih)
  gemm_wmma<4><<<dim3(64, 128), b32, 0, stream>>>(X2, l1_wih, l1_b, BIG, 256, 768, 768, 4096, 0);
  lstm1_act<<<8192, 256, 0, stream>>>(BIG, Q);
  // Attention precomputes + scan
  gemm_wmma<4><<<dim3(2, 192), b32, 0, stream>>>(enc, wk, nullptr, KEYS, 512, 512, 512, 128, 0);
  gemm_wmma<4><<<dim3(2, 128), b32, 0, stream>>>(Q, wq, nullptr, QP, 1024, 1024, 1024, 128, 0);
  attention_kernel<<<16, 256, 0, stream>>>(enc, KEYS, QP, locw, wloc, vvec, X2, DEC);
  // LSTM2: batched input-gates GEMM, then WMMA recurrence
  gemm_wmma<4><<<dim3(64, 128), b32, 0, stream>>>(X2, l2_wih, l2_b, BIG, 768, 768, 768, 4096, 0);
  cvt_f16<<<16384, 256, 0, stream>>>(l2_whh, WHH, 4096 * 1024);
  lstm2_kernel<<<1, 1024, 96 * 1024, stream>>>(BIG, WHH, DEC);
  // Output projection (N=80 -> one 5-tile block column)
  gemm_wmma<5><<<dim3(1, 128), b32, 0, stream>>>(DEC, out_w, out_b, frames, 1536, 1536, 1536, 80, 0);
  // PostNet
  im2col<<<(2048 * 400 + 255) / 256, 256, 0, stream>>>(frames, BIG, 80);
  gemm_wmma<4><<<dim3(8, 128), b32, 0, stream>>>(BIG, pnw[0], nullptr, P, 400, 400, 400, 512, 0);
  bn_stats<<<512, 256, 0, stream>>>(P, 512, png[0], pnb[0], SC, SH);
  bn_apply<<<4096, 256, 0, stream>>>(P, 512, SC, SH, 1);
  for (int l = 1; l < 4; ++l) {
    im2col<<<(2048 * 2560 + 255) / 256, 256, 0, stream>>>(P, BIG, 512);
    gemm_wmma<4><<<dim3(8, 128), b32, 0, stream>>>(BIG, pnw[l], nullptr, P, 2560, 2560, 2560, 512, 0);
    bn_stats<<<512, 256, 0, stream>>>(P, 512, png[l], pnb[l], SC, SH);
    bn_apply<<<4096, 256, 0, stream>>>(P, 512, SC, SH, 1);
  }
  im2col<<<(2048 * 2560 + 255) / 256, 256, 0, stream>>>(P, BIG, 512);
  gemm_wmma<5><<<dim3(1, 128), b32, 0, stream>>>(BIG, pnw[4], nullptr, R, 2560, 2560, 2560, 80, 0);
  bn_stats<<<80, 256, 0, stream>>>(R, 80, png[4], pnb[4], SC, SH);
  residual_out<<<640, 256, 0, stream>>>(frames, R, SC, SH, out2);
}
